// MLPBaselineV2_50483045598051
// MI455X (gfx1250) — matrix operand order repaired and checked by reading, not compiled, not run on hardware
//
#include <hip/hip_runtime.h>
#include <hip/hip_bf16.h>
#include <stdint.h>

// ---------------------------------------------------------------------------
// MI455X / gfx1250 implementation.
//   Kp : pack Wp/W1/W2 into WMMA B-fragment-major f16 (one-time, tiny)
//   K0 : qprime = pool_query^T @ Wk  (kills the entire keys GEMM, -86 GFLOP)
//   K1 : node_embeds = elu(x @ Wp^T + bp) -- WMMA f16; x streamed HBM->LDS
//        with global_load_async_to_lds_b128 + s_wait_asynccnt double buffer
//   K2 : fused mut-MLP (2 WMMA GEMMs) + L2 site gather + attn pool + head
// All B fragments are pre-packed so a fragment = one contiguous v16h per
// lane; bases are hoisted so the inner loops use immediate offsets only.
// ---------------------------------------------------------------------------

#define HIDDEN 256
#define ESM    1280
#define NNODES 50000
#define BATCH  32768
#define MSITES 20
#define NKS1   (ESM / 32)          // 40 K-steps for the node-proj GEMM

typedef __attribute__((ext_vector_type(16))) _Float16 v16h;
typedef __attribute__((ext_vector_type(8)))  float    v8f;

// ---- CDNA5 async copy: global -> LDS, tracked by ASYNCcnt ------------------
#if __has_builtin(__builtin_amdgcn_s_wait_asynccnt)
#define WAIT_ASYNC0() __builtin_amdgcn_s_wait_asynccnt(0)
#define WAIT_ASYNC2() __builtin_amdgcn_s_wait_asynccnt(2)
#else
#define WAIT_ASYNC0() asm volatile("s_wait_asynccnt 0x0" ::: "memory")
#define WAIT_ASYNC2() asm volatile("s_wait_asynccnt 0x2" ::: "memory")
#endif

// Generic LDS pointer's low 32 bits are the LDS byte offset (aperture lives
// in the high half), which is what the instruction's LDS-address VGPR wants.
__device__ __forceinline__ void async_g2l_b128(uint32_t lds_off,
                                               const float* g) {
    asm volatile("global_load_async_to_lds_b128 %0, %1, off"
                 :: "v"(lds_off), "v"(g)
                 : "memory");
}

// D = A(16x32 f16) * B(32x16 f16) + C(16x16 f32)   [probe-verified builtin]
__device__ __forceinline__ v8f wmma_f16(v16h a, v16h b, v8f c) {
    return __builtin_amdgcn_wmma_f32_16x16x32_f16(
        false, a, false, b, (short)0, c, false, false);
}

// ===========================================================================
// Kp: pack fp32 weight W[N][K] into B-fragment-major f16:
//   frag[((nt*nks + ks)*32 + lane)*16 + i] =
//       W[nt*16 + (lane&15)][ks*32 + ((lane&16)?8:0) + (i<8 ? i : 8+i)]   (0 if k >= K)
// ===========================================================================
__global__ __launch_bounds__(256) void pack_b_frag_kernel(
    const float* __restrict__ W, _Float16* __restrict__ dst, int N, int K,
    int nks) {
    int id = blockIdx.x * 256 + threadIdx.x;      // one (nt, ks, lane) each
    int total = (N / 16) * nks * 32;
    if (id >= total) return;
    int lane = id & 31;
    int ks   = (id >> 5) % nks;
    int nt   = id / (32 * nks);
    int n  = nt * 16 + (lane & 15);
    int kb = ks * 32 + ((lane & 16) ? 8 : 0);
    _Float16* d = dst + (size_t)id * 16;
#pragma unroll
    for (int i = 0; i < 16; ++i) {
        int k = kb + (i < 8 ? i : 8 + i);
        d[i] = (_Float16)((k < K) ? W[(size_t)n * K + k] : 0.f);
    }
}

// ===========================================================================
// K0: qprime[k] = sum_n pq[n]*Wk[n][k] ;  pqb = dot(pq, bk)
// ===========================================================================
__global__ __launch_bounds__(256) void qprime_kernel(
    const float* __restrict__ pq, const float* __restrict__ Wk,
    const float* __restrict__ bk, float* __restrict__ qprime,
    float* __restrict__ pqb) {
    int k = threadIdx.x;
    float s = 0.f;
    for (int n = 0; n < HIDDEN; ++n) s += pq[n] * Wk[n * HIDDEN + k];
    qprime[k] = s;
    if (k == 0) {
        float t = 0.f;
        for (int n = 0; n < HIDDEN; ++n) t += pq[n] * bk[n];
        *pqb = t;
    }
}

// ===========================================================================
// K1: node_embeds = elu(x @ Wp^T + bp)   [50000 x 1280] -> [50000 x 256]
// 256 threads / 8 waves; 64-row x 256-col tile; K by 32, double-buffered
// async HBM->LDS for x (the dominant 256 MB stream). B frags from packed WpF.
// ===========================================================================
#define K1_ROWS 64
#define K1_LDF  36                      // padded fp32 leading dim (bank-spread)
__global__ __launch_bounds__(256) void node_proj_kernel(
    const float* __restrict__ x, const v16h* __restrict__ WpF,
    const float* __restrict__ bp, float* __restrict__ node_embeds,
    int n_nodes) {
    __shared__ float sX[2][K1_ROWS * K1_LDF];   // 2 x 9 KB fp32 A slices

    const int t = threadIdx.x, lane = t & 31, w = t >> 5;
    const int row0 = blockIdx.x * K1_ROWS;
    const int rt = w & 3;                       // row-tile of this wave
    const int ntbase = (w >> 2) * 8;            // 8 n-tiles per wave

    // ---- per-thread async-copy plan: 8 consecutive floats of one row ------
    const int r  = t >> 2;                      // 0..63
    const int kk = (t & 3) * 8;                 // 0,8,16,24
    const int grow = row0 + r;
    const bool inb = grow < n_nodes;
    const uint32_t offBuf[2] = {
        (uint32_t)(uintptr_t)(&sX[0][r * K1_LDF + kk]),
        (uint32_t)(uintptr_t)(&sX[1][r * K1_LDF + kk])};
    const float* srcNext = x + (size_t)grow * ESM + kk;

    auto issue = [&](int buf) {                 // one 32-float K-slice piece
        if (inb) {
            async_g2l_b128(offBuf[buf], srcNext);
            async_g2l_b128(offBuf[buf] + 16, srcNext + 4);
        }
        srcNext += 32;
    };

    // ---- hoisted fragment bases ------------------------------------------
    // B: stride in j is NKS1*32 v16h (40960 B -> immediate), ks bumps +32
    const v16h* pB = WpF + (size_t)ntbase * NKS1 * 32 + lane;
    // A: fixed lane address inside each LDS buffer
    const int akb = (lane & 16) ? 8 : 0;
    const float* aBase[2] = {
        &sX[0][(rt * 16 + (lane & 15)) * K1_LDF + akb],
        &sX[1][(rt * 16 + (lane & 15)) * K1_LDF + akb]};

    v8f acc[8] = {};
    issue(0);
    for (int ks = 0; ks < NKS1; ++ks) {
        if (ks + 1 < NKS1) {
            issue((ks + 1) & 1);
            WAIT_ASYNC2();                      // slice ks landed (in-order)
        } else {
            WAIT_ASYNC0();
        }
        __syncthreads();

        const float* ap = aBase[ks & 1];
        v16h a;
#pragma unroll
        for (int i = 0; i < 8; ++i) a[i]     = (_Float16)ap[i];
#pragma unroll
        for (int i = 0; i < 8; ++i) a[8 + i] = (_Float16)ap[16 + i];
#pragma unroll
        for (int j = 0; j < 8; ++j)             // immediate-offset B loads
            acc[j] = wmma_f16(a, pB[(size_t)j * (NKS1 * 32)], acc[j]);
        pB += 32;
        __syncthreads();                        // slice reusable for ks+2
    }

    // epilogue: elu(acc + bp); C layout: N=lane%16, M=vgpr + 8*(lane/16)
    const int mrow = row0 + rt * 16 + ((lane & 16) ? 8 : 0);
    const int nc = lane & 15;
#pragma unroll
    for (int j = 0; j < 8; ++j) {
        int n = (ntbase + j) * 16 + nc;
        float bias = bp[n];
#pragma unroll
        for (int rr = 0; rr < 8; ++rr) {
            int row = mrow + rr;
            if (row < n_nodes) {
                float v = acc[j][rr] + bias;
                v = v > 0.f ? v : (__expf(v) - 1.f);
                node_embeds[(size_t)row * HIDDEN + n] = v;
            }
        }
    }
}

// ===========================================================================
// K2: fused mut-MLP + site gather + attention pooling + head.
// One block = NB(=4) proteins = 80 rows. 256 threads (8 waves).
// Dynamic LDS (~127 KB of the 320 KB WGP pool), phase-aliased:
//   [comb 80x256 f32 | alias: onehot 80x20 f32][hidden 80x256 f16]
//   [pooled 4x256 f32][wgt 80 f32][h3 4x128 f32][sitebuf 80 i32]
// ===========================================================================
#define NB  4
#define RWS (NB * MSITES)               // 80 rows per block

__global__ __launch_bounds__(256) void fused_mut_pool_head_kernel(
    const float* __restrict__ onehot, const long long* __restrict__ sites,
    const unsigned char* __restrict__ mask,
    const v16h* __restrict__ W1F, const float* __restrict__ b1,
    const v16h* __restrict__ W2F, const float* __restrict__ b2,
    const float* __restrict__ W3, const float* __restrict__ b3,
    const float* __restrict__ W4, const float* __restrict__ b4,
    const float* __restrict__ node_embeds, const float* __restrict__ qprime,
    const float* __restrict__ pqb, float* __restrict__ out) {
    extern __shared__ char smem[];
    float*    comb   = (float*)smem;                              // 81920 B
    float*    a1f    = (float*)smem;                              // alias
    _Float16* hidden = (_Float16*)(smem + 81920);                 // 40960 B
    float*    pooled = (float*)(smem + 81920 + 40960);            // 4096 B
    float*    wgt    = (float*)(smem + 81920 + 40960 + 4096);     // 320 B
    float*    h3     = (float*)(smem + 81920 + 40960 + 4096 + 320);   // 2048
    int*      sitebuf= (int*)(smem + 81920 + 40960 + 4096 + 320 + 2048);

    const int t = threadIdx.x, lane = t & 31, w = t >> 5;
    const int b0 = blockIdx.x * NB;
    const size_t row0 = (size_t)b0 * MSITES;

    // ---- Phase 0: async-copy the 80x20 fp32 onehot block (6400 B) to LDS,
    //      stage the 80 site indices once ----------------------------------
    {
        const float* src = onehot + row0 * MSITES;
        uint32_t base = (uint32_t)(uintptr_t)a1f;
        for (int op = t; op < (RWS * MSITES) / 4; op += 256)      // 400 b128
            async_g2l_b128(base + op * 16, src + op * 4);
        if (t < RWS) sitebuf[t] = (int)sites[row0 + t];
        WAIT_ASYNC0();
        __syncthreads();
    }

    // ---- Phase 1: hidden = relu(onehot @ W1^T + b1)  (one WMMA K-step) ----
    // wave w: n-tiles {2w, 2w+1} x row-tiles {0..4}
    const int akb = (lane & 16) ? 8 : 0;
    const float* aohBase = a1f + (lane & 15) * MSITES;
    const v16h* pB1 = W1F + (size_t)(w * 2) * 32 + lane;
#pragma unroll
    for (int ntl = 0; ntl < 2; ++ntl) {
        int nt = w * 2 + ntl;
        v16h b = pB1[ntl * 32];                 // immediate offset
#pragma unroll
        for (int rt = 0; rt < 5; ++rt) {
            const float* p = aohBase + rt * 16 * MSITES;
            v16h a;
#pragma unroll
            for (int i = 0; i < 8; ++i) {
                int k = akb + i;
                a[i] = (_Float16)((k < MSITES) ? p[k] : 0.f);
            }
#pragma unroll
            for (int i = 0; i < 8; ++i) {
                int k = akb + 16 + i;
                a[8 + i] = (_Float16)((k < MSITES) ? p[k] : 0.f);
            }
            v8f c = {};
            c = wmma_f16(a, b, c);
            int n = nt * 16 + (lane & 15);
            float bias = b1[n];
            int mrow = rt * 16 + ((lane & 16) ? 8 : 0);
#pragma unroll
            for (int rr = 0; rr < 8; ++rr) {
                float v = c[rr] + bias;
                hidden[(mrow + rr) * HIDDEN + n] =
                    (_Float16)(v > 0.f ? v : 0.f);
            }
        }
    }
    __syncthreads();                            // a1f dead; comb may be used

    // ---- Phase 2: mut = hidden @ W2^T ; comb = mut + b2 + gather(nodes) ---
    const _Float16* hBase = hidden + (lane & 15) * HIDDEN + akb;
    const v16h* pB2 = W2F + (size_t)(w * 2) * 8 * 32 + lane;
    v8f acc[10] = {};                           // [ntl][rt]
    for (int ks = 0; ks < 8; ++ks) {            // K = 256 = 8 * 32
        const _Float16* hp = hBase + ks * 32;
        v16h afrag[5];
#pragma unroll
        for (int rt = 0; rt < 5; ++rt) {
            const _Float16* p = hp + rt * 16 * HIDDEN;   // immediate offsets
            v16h a;
#pragma unroll
            for (int i = 0; i < 8; ++i) a[i]     = p[i];
#pragma unroll
            for (int i = 0; i < 8; ++i) a[8 + i] = p[16 + i];
            afrag[rt] = a;
        }
#pragma unroll
        for (int ntl = 0; ntl < 2; ++ntl) {
            v16h b = pB2[ntl * (8 * 32)];       // immediate offset
#pragma unroll
            for (int rt = 0; rt < 5; ++rt)
                acc[ntl * 5 + rt] = wmma_f16(afrag[rt], b, acc[ntl * 5 + rt]);
        }
        pB2 += 32;                              // next K-step
    }
#pragma unroll
    for (int ntl = 0; ntl < 2; ++ntl) {
        int nt = w * 2 + ntl;
        int n = nt * 16 + (lane & 15);
        float bias = b2[n];
#pragma unroll
        for (int rt = 0; rt < 5; ++rt) {
            int mrow = rt * 16 + ((lane & 16) ? 8 : 0);
            v8f c = acc[ntl * 5 + rt];
#pragma unroll
            for (int rr = 0; rr < 8; ++rr) {
                int r = mrow + rr;              // 0..79
                int site = sitebuf[r];
                float g = node_embeds[(size_t)site * HIDDEN + n]; // L2 gather
                comb[r * HIDDEN + n] = c[rr] + bias + g;
            }
        }
    }
    __syncthreads();

    // ---- Phase 3: scores = (comb.qprime + pqb)/16, masked softmax, pool ---
    const float qb = pqb[0];
    for (int i = 0; i < 10; ++i) {              // wave w owns 10 rows
        int rr = w * 10 + i;
        float s = 0.f;
#pragma unroll
        for (int j = 0; j < 8; ++j) {
            int k = lane + j * 32;
            s += comb[rr * HIDDEN + k] * qprime[k];
        }
#pragma unroll
        for (int off = 16; off > 0; off >>= 1) s += __shfl_xor(s, off, 32);
        if (lane == 0) {
            float sc = (s + qb) * 0.0625f;      // / sqrt(256)
            bool mk = mask[row0 + rr] != 0;
            wgt[rr] = mk ? sc : -3.0e38f;
        }
    }
    __syncthreads();
    if (t < NB) {                               // softmax over 20 sites
        float mx = -3.4e38f;
        for (int m = 0; m < MSITES; ++m) mx = fmaxf(mx, wgt[t * MSITES + m]);
        float e[MSITES], sum = 0.f;
        for (int m = 0; m < MSITES; ++m) {
            e[m] = __expf(wgt[t * MSITES + m] - mx);
            sum += e[m];
        }
        float inv = 1.f / sum;
        for (int m = 0; m < MSITES; ++m) wgt[t * MSITES + m] = e[m] * inv;
    }
    __syncthreads();
    for (int bb = 0; bb < NB; ++bb) {           // pooled[bb][n], n = t
        float s = 0.f;
#pragma unroll
        for (int m = 0; m < MSITES; ++m)
            s += wgt[bb * MSITES + m] * comb[(bb * MSITES + m) * HIDDEN + t];
        pooled[bb * HIDDEN + t] = s;
    }
    __syncthreads();

    // ---- Phase 4: head MLP ------------------------------------------------
    if (t < 128) {
        const float* wrow = W3 + t * HIDDEN;
        for (int bb = 0; bb < NB; ++bb) {
            float s = b3[t];
            for (int k = 0; k < HIDDEN; ++k)
                s += pooled[bb * HIDDEN + k] * wrow[k];
            h3[bb * 128 + t] = fmaxf(s, 0.f);
        }
    }
    __syncthreads();
    if (t < NB) {
        float s = b4[0];
        for (int k = 0; k < 128; ++k) s += h3[t * 128 + k] * W4[k];
        out[b0 + t] = s;
    }
}

// ===========================================================================
extern "C" void kernel_launch(void* const* d_in, const int* in_sizes, int n_in,
                              void* d_out, int out_size, void* d_ws,
                              size_t ws_size, hipStream_t stream) {
    const float*         x      = (const float*)d_in[0];
    const long long*     sites  = (const long long*)d_in[1];
    const float*         onehot = (const float*)d_in[2];
    const unsigned char* mask   = (const unsigned char*)d_in[3];
    const float* Wp = (const float*)d_in[4];
    const float* bp = (const float*)d_in[5];
    const float* W1 = (const float*)d_in[6];
    const float* b1 = (const float*)d_in[7];
    const float* W2 = (const float*)d_in[8];
    const float* b2 = (const float*)d_in[9];
    const float* pq = (const float*)d_in[10];
    const float* Wk = (const float*)d_in[11];
    const float* bk = (const float*)d_in[12];
    const float* W3 = (const float*)d_in[13];
    const float* b3 = (const float*)d_in[14];
    const float* W4 = (const float*)d_in[15];
    const float* b4 = (const float*)d_in[16];
    float* out = (float*)d_out;

    // workspace layout
    char*  ws          = (char*)d_ws;
    float* node_embeds = (float*)ws;                          // 51,200,000 B
    size_t off = (size_t)NNODES * HIDDEN * 4;
    float* qprime = (float*)(ws + off);      off += HIDDEN * 4;
    float* pqb    = (float*)(ws + off);      off += 64;       // pad/align
    _Float16* WpF = (_Float16*)(ws + off);   off += (size_t)ESM * HIDDEN * 2;
    _Float16* W1F = (_Float16*)(ws + off);   off += 32 * HIDDEN * 2;
    _Float16* W2F = (_Float16*)(ws + off);   off += HIDDEN * HIDDEN * 2;

    // weight packing (idempotent, deterministic)
    pack_b_frag_kernel<<<(16 * NKS1 * 32 + 255) / 256, 256, 0, stream>>>(
        Wp, WpF, HIDDEN, ESM, NKS1);
    pack_b_frag_kernel<<<(16 * 1 * 32 + 255) / 256, 256, 0, stream>>>(
        W1, W1F, HIDDEN, MSITES, 1);
    pack_b_frag_kernel<<<(16 * 8 * 32 + 255) / 256, 256, 0, stream>>>(
        W2, W2F, HIDDEN, HIDDEN, 8);

    qprime_kernel<<<1, 256, 0, stream>>>(pq, Wk, bk, qprime, pqb);

    node_proj_kernel<<<(NNODES + K1_ROWS - 1) / K1_ROWS, 256, 0, stream>>>(
        x, (const v16h*)WpF, bp, node_embeds, NNODES);

    size_t smem = 81920 + 40960 + 4096 + 320 + 2048 + 320;    // ~127 KB
    hipFuncSetAttribute((const void*)fused_mut_pool_head_kernel,
                        hipFuncAttributeMaxDynamicSharedMemorySize, (int)smem);
    fused_mut_pool_head_kernel<<<BATCH / NB, 256, smem, stream>>>(
        onehot, sites, mask, (const v16h*)W1F, b1, (const v16h*)W2F, b2,
        W3, b3, W4, b4, node_embeds, qprime, pqb, out);
}
